// TransitionNER_20830591385703
// MI455X (gfx1250) — compile-verified, matrix operand order checked by reading, not run on hardware
//
#include <hip/hip_runtime.h>
#include <hip/hip_bf16.h>

// ---------------------------------------------------------------------------
// TransitionNER fused pipeline for gfx1250 (MI455X):
//   k0: W1 fp32 [4096,1024] -> bf16 transposed W1T [1024][4096]  (L2-resident)
//   k1: hidden = tanh(concat(X4) @ W1 + b1), double-buffered LDS,
//       X staged via regs (fp32->bf16 via v_perm_b32 pack), W1T via
//       global_load_async_to_lds, compute with v_wmma_f32_16x16x32_bf16
//   k2: logits = hidden @ W2 + b2; masked log-softmax; argmax + gather loss
// ---------------------------------------------------------------------------

typedef __attribute__((ext_vector_type(16))) __bf16 v16bf;
typedef __attribute__((ext_vector_type(8)))  float  v8f;

union Frag { uint4 u[2]; v16bf v; };

__device__ __forceinline__ unsigned short f2bf(float f) {
  unsigned int u = __builtin_bit_cast(unsigned int, f);
  u += 0x7FFFu + ((u >> 16) & 1u);          // round-to-nearest-even
  return (unsigned short)(u >> 16);
}
// pack two fp32 -> packed bf16 pair.
//  - prefer hw packed convert if the toolchain declares it
//  - else: half-ulp bias + single v_perm_b32 picking the two high halves
//    (3 VALU per 2 elements, no 16-bit subregister moves)
__device__ __forceinline__ unsigned int pack2(float a, float b) {
#if __has_builtin(__builtin_amdgcn_cvt_pk_bf16_f32)
  return __builtin_bit_cast(unsigned int, __builtin_amdgcn_cvt_pk_bf16_f32(a, b));
#else
  unsigned int ua = __builtin_bit_cast(unsigned int, a) + 0x8000u;
  unsigned int ub = __builtin_bit_cast(unsigned int, b) + 0x8000u;
  // result bytes: [0]=ua[2],[1]=ua[3],[2]=ub[2],[3]=ub[3]
  return __builtin_amdgcn_perm(ub, ua, 0x07060302u);
#endif
}
__device__ __forceinline__ float bf2f(unsigned short h) {
  return __builtin_bit_cast(float, (unsigned int)h << 16);
}
// branch-free tanh: v_exp_f32 + fast rcp, no libm control flow
__device__ __forceinline__ float fast_tanh(float x) {
  float ax = __builtin_fabsf(x);
  float e  = __expf(-2.0f * ax);
  float r  = __fdividef(1.0f - e, 1.0f + e);
  return __builtin_copysignf(r, x);
}

// ---------------------------------------------------------------- kernel 0 --
__global__ __launch_bounds__(256) void transpose_w1_bf16(
    const float* __restrict__ W1, unsigned short* __restrict__ w1t) {
  int idx = blockIdx.x * 256 + threadIdx.x;     // 4096*1024 elements
  int k = idx >> 10, n = idx & 1023;            // W1 is [K=4096][N=1024]
  w1t[(size_t)n * 4096 + k] = f2bf(W1[idx]);    // W1T is [N=1024][K=4096]
}

// ---------------------------------------------------------------- kernel 1 --
// Workgroup tile 128(M) x 128(N), K staged 64/iter, double-buffered LDS.
// 8 waves 4(M) x 2(N); each wave computes 32x64 via 2x4 WMMA tiles.
#define LDSW 36                 // dwords per LDS row: 32 data (64 bf16) + 4 pad
#define BUFSZ (128 * LDSW)      // dwords per buffer

__global__ __launch_bounds__(256) void gemm1_tanh(
    const float* __restrict__ bufh, const float* __restrict__ stkh,
    const float* __restrict__ outh, const float* __restrict__ acth,
    const unsigned short* __restrict__ w1t,    // bf16 W1T [1024][4096]
    const float* __restrict__ b1,
    unsigned short* __restrict__ hidden)       // bf16 [16384][1024]
{
  __shared__ __align__(16) unsigned int sA[2 * BUFSZ];
  __shared__ __align__(16) unsigned int sB[2 * BUFSZ];

  const int t  = threadIdx.x;
  const int n0 = blockIdx.x * 128;   // N-groups fastest -> X slab reuse in L2
  const int m0 = blockIdx.y * 128;
  const int wave = t >> 5, lane = t & 31;
  const int wm = wave >> 1, wn = wave & 1;
  const int l15 = lane & 15, hi = lane >> 4;
  // per-thread staging coordinates (invariant across K blocks)
  const int xm = t >> 4, xkq = t & 15;   // X: 8 float4, rows xm+16i, col xkq
  const int wn8 = t >> 3, wkq = t & 7;   // W: 4 uint4,  rows wn8+32i, col wkq

  v8f acc[2][4] = {};
  float4 xr[8];

  // issue X global loads for K-block kb_ into registers
#define ISSUE_X(kb_)                                                        \
  {                                                                         \
    const float* src = (kb_) < 1024 ? bufh : (kb_) < 2048 ? stkh            \
                     : (kb_) < 3072 ? outh : acth;                          \
    const int koff = (kb_) & 1023;                                          \
    _Pragma("unroll")                                                       \
    for (int i = 0; i < 8; ++i)                                             \
      xr[i] = *(const float4*)(src + (size_t)(m0 + xm + 16 * i) * 1024      \
                               + koff + xkq * 4);                           \
  }

  // async DMA W1T tile for K-block kb_ straight into LDS buffer buf_
#define ISSUE_W_ASYNC(kb_, buf_)                                            \
  {                                                                         \
    _Pragma("unroll")                                                       \
    for (int i = 0; i < 4; ++i) {                                           \
      int n = wn8 + 32 * i;                                                 \
      unsigned long long ga = (unsigned long long)w1t +                     \
          ((size_t)(n0 + n) * 4096 + (kb_)) * 2 + (size_t)wkq * 16;         \
      unsigned lo = (unsigned)(size_t)&sB[(buf_) * BUFSZ + n * LDSW + wkq * 4]; \
      asm volatile("global_load_async_to_lds_b128 %0, %1, off"              \
                   :: "v"(lo), "v"(ga) : "memory");                         \
    }                                                                       \
  }

  // convert (v_perm pack) + store staged X registers into buffer buf_
#define STORE_X(buf_)                                                       \
  {                                                                         \
    _Pragma("unroll")                                                       \
    for (int i = 0; i < 8; ++i)                                             \
      *(uint2*)&sA[(buf_) * BUFSZ + (xm + 16 * i) * LDSW + xkq * 2] =       \
          make_uint2(pack2(xr[i].x, xr[i].y), pack2(xr[i].z, xr[i].w));     \
  }

  // ---- prologue: stage block 0 into buffer 0
  ISSUE_X(0)
  ISSUE_W_ASYNC(0, 0)
  STORE_X(0)
  asm volatile("s_wait_asynccnt 0x0" ::: "memory");
  __syncthreads();

  for (int ib = 0; ib < 64; ++ib) {
    const int cur = ib & 1, nxt = cur ^ 1;
    const bool more = (ib + 1) < 64;
    if (more) {
      const int kb = (ib + 1) * 64;
      ISSUE_X(kb)             // global loads overlap with WMMA below
      ISSUE_W_ASYNC(kb, nxt)  // async DMA into other buffer
    }

    // ---- compute on current buffer: two k=32 steps, all fragments up front
    const int abase = cur * BUFSZ;
    #pragma unroll
    for (int kk = 0; kk < 2; ++kk) {
      // ISA 16-bit fragment layout: lanes 0-15 take K0..7,K16..23;
      // lanes 16-31 take K8..15,K24..31 -> two ds_load_b128 per fragment.
      const int kbase = kk * 16 + hi * 4;
      Frag a0, a1, b[4];
      {
        int ra = abase + (wm * 32 + l15) * LDSW + kbase;
        a0.u[0] = *(const uint4*)&sA[ra];
        a0.u[1] = *(const uint4*)&sA[ra + 8];
        int rb = abase + (wm * 32 + 16 + l15) * LDSW + kbase;
        a1.u[0] = *(const uint4*)&sA[rb];
        a1.u[1] = *(const uint4*)&sA[rb + 8];
      }
      #pragma unroll
      for (int jn = 0; jn < 4; ++jn) {
        int rb = abase + (wn * 64 + jn * 16 + l15) * LDSW + kbase;
        b[jn].u[0] = *(const uint4*)&sB[rb];
        b[jn].u[1] = *(const uint4*)&sB[rb + 8];
      }
      #pragma unroll
      for (int jn = 0; jn < 4; ++jn) {
        acc[0][jn] = __builtin_amdgcn_wmma_f32_16x16x32_bf16(
            false, a0.v, false, b[jn].v, (short)0, acc[0][jn], false, false);
        acc[1][jn] = __builtin_amdgcn_wmma_f32_16x16x32_bf16(
            false, a1.v, false, b[jn].v, (short)0, acc[1][jn], false, false);
      }
    }

    if (more) STORE_X(nxt)
    asm volatile("s_wait_asynccnt 0x0" ::: "memory");
    __syncthreads();
  }

  // ---- epilogue: + b1, tanh, store bf16 hidden
  // C/D layout: lane (0..15) = N, +16 lanes -> M+8; VGPR j -> M row j.
  // Convert two M-rows per pack2, store halves.
  #pragma unroll
  for (int jn = 0; jn < 4; ++jn) {
    int n = n0 + wn * 64 + jn * 16 + l15;
    float bias = b1[n];
    #pragma unroll
    for (int i = 0; i < 2; ++i) {
      size_t base = (size_t)(m0 + wm * 32 + i * 16 + hi * 8) * 1024 + n;
      #pragma unroll
      for (int j = 0; j < 8; j += 2) {
        unsigned p = pack2(fast_tanh(acc[i][jn][j] + bias),
                           fast_tanh(acc[i][jn][j + 1] + bias));
        hidden[base + (size_t)j * 1024]       = (unsigned short)p;
        hidden[base + (size_t)(j + 1) * 1024] = (unsigned short)(p >> 16);
      }
    }
  }
#undef ISSUE_X
#undef ISSUE_W_ASYNC
#undef STORE_X
}

// ---------------------------------------------------------------- kernel 2 --
// 128 rows/block. logits accumulate in registers (2 threads/row x 10 acts),
// then masked log-softmax, argmax (first-max), gold-action gather.
__global__ __launch_bounds__(256) void head_logsoftmax(
    const unsigned short* __restrict__ hidden,  // bf16 [B][1024]
    const float* __restrict__ W2,               // [1024][20]
    const float* __restrict__ b2,               // [20]
    const unsigned char* __restrict__ valid,    // bool [B][20]
    const int* __restrict__ realAct,            // [B]
    float* __restrict__ out, int B)
{
  __shared__ __align__(16) unsigned int sH[128 * 36];  // 128 rows x 64 bf16 (+pad)
  __shared__ float sW2[64 * 20];
  __shared__ float sLog[128][20];

  const int t = threadIdx.x;
  const int r0 = blockIdx.x * 128;
  const int row = t >> 1, a0 = (t & 1) * 10;

  float accv[10];
  #pragma unroll
  for (int j = 0; j < 10; ++j) accv[j] = 0.f;

  for (int kc = 0; kc < 1024; kc += 64) {
    const uint4* h4 = (const uint4*)hidden;
    #pragma unroll
    for (int i = 0; i < 4; ++i) {
      int g = t + i * 256;
      int r = g >> 3, kq = g & 7;
      *(uint4*)&sH[r * 36 + kq * 4] = h4[(size_t)(r0 + r) * 128 + (kc >> 3) + kq];
    }
    for (int i = t; i < 64 * 20; i += 256)
      sW2[i] = W2[(size_t)(kc + i / 20) * 20 + (i % 20)];
    __syncthreads();

    const unsigned short* sHs = (const unsigned short*)sH;
    for (int k = 0; k < 64; ++k) {
      float h = bf2f(sHs[row * 72 + k]);
      #pragma unroll
      for (int j = 0; j < 10; ++j)
        accv[j] = fmaf(h, sW2[k * 20 + a0 + j], accv[j]);
    }
    __syncthreads();
  }

  #pragma unroll
  for (int j = 0; j < 10; ++j) sLog[row][a0 + j] = accv[j];
  __syncthreads();

  if (t < 128) {
    int gr = r0 + t;
    float masked[20];
    float mx = -3.4e38f;
    int arg = 0;
    #pragma unroll
    for (int a = 0; a < 20; ++a) {
      float v = sLog[t][a] + b2[a];
      if (!valid[(size_t)gr * 20 + a]) v = -1e9f;
      masked[a] = v;
      if (v > mx) { mx = v; arg = a; }   // first occurrence wins (jnp.argmax)
    }
    float s = 0.f;
    #pragma unroll
    for (int a = 0; a < 20; ++a) s += __expf(masked[a] - mx);
    float logZ = mx + __logf(s);
    out[gr]     = (float)arg;                 // predict_actions (as float)
    out[B + gr] = masked[realAct[gr]] - logZ; // losses
  }
}

// ------------------------------------------------------------------ launch --
extern "C" void kernel_launch(void* const* d_in, const int* in_sizes, int n_in,
                              void* d_out, int out_size, void* d_ws, size_t ws_size,
                              hipStream_t stream) {
  const float* bufh = (const float*)d_in[0];
  const float* stkh = (const float*)d_in[1];
  const float* outh = (const float*)d_in[2];
  const float* acth = (const float*)d_in[3];
  const float* W1   = (const float*)d_in[4];
  const float* b1   = (const float*)d_in[5];
  const float* W2   = (const float*)d_in[6];
  const float* b2   = (const float*)d_in[7];
  const unsigned char* valid = (const unsigned char*)d_in[8];
  const int* realAct = (const int*)d_in[9];
  float* out = (float*)d_out;

  const int B = 16384, H = 1024;

  // workspace: W1T bf16 (8 MB) | hidden bf16 (32 MB)
  char* ws = (char*)d_ws;
  unsigned short* w1t    = (unsigned short*)ws;
  unsigned short* hidden = (unsigned short*)(ws + (size_t)H * (4 * H) * sizeof(unsigned short));

  transpose_w1_bf16<<<(4 * H * H) / 256, 256, 0, stream>>>(W1, w1t);

  dim3 g1(H / 128, B / 128);   // x = N-group (8), y = M-group (128): L2 reuse of X
  gemm1_tanh<<<g1, 256, 0, stream>>>(bufh, stkh, outh, acth, w1t, b1, hidden);

  head_logsoftmax<<<B / 128, 256, 0, stream>>>(hidden, W2, b2, valid, realAct, out, B);
}
